// basket_GRU_MI2_76295799046564
// MI455X (gfx1250) — matrix operand order repaired and checked by reading, not compiled, not run on hardware
//
#include <hip/hip_runtime.h>
#include <hip/hip_bf16.h>

typedef _Float16 half_t;
typedef __attribute__((ext_vector_type(16))) _Float16 v16h;
typedef __attribute__((ext_vector_type(8)))  _Float16 v8h;
typedef __attribute__((ext_vector_type(8)))  float    v8f;

#define B_   256
#define T_   200
#define I_   512
#define H_   256
#define NROW (B_ * T_)   // 51200

static __device__ __forceinline__ float sigmoidf_(float x) {
  return 1.0f / (1.0f + __expf(-x));
}

// ---------------------------------------------------------------------------
// A-fragment loaders (16x32 f16 tile, ISA 7.12.2 layout):
//   lane m (m=lane&15, sel=lane>>4): halves 0..7  -> K = k0 + sel*8 + j
//                                    halves 8..15 -> K = k0 + 16 + sel*8 + j
// ---------------------------------------------------------------------------
static __device__ __forceinline__ v16h a_frag_f32(const float* __restrict__ src,
                                                  long ld, int row0, int k0, int lane) {
  const int m = lane & 15, sel = lane >> 4;
  const float* p = src + (long)(row0 + m) * ld + k0 + sel * 8;
  v16h a;
#pragma unroll
  for (int j = 0; j < 8; ++j) a[j] = (_Float16)p[j];
#pragma unroll
  for (int j = 0; j < 8; ++j) a[8 + j] = (_Float16)p[16 + j];
  return a;
}

static __device__ __forceinline__ v16h a_frag_f16(const half_t* __restrict__ src,
                                                  long ld, int row0, int k0, int lane) {
  const int m = lane & 15, sel = lane >> 4;
  const half_t* p = src + (long)(row0 + m) * ld + k0 + sel * 8;
  v16h a;
#pragma unroll
  for (int j = 0; j < 8; ++j) a[j] = p[j];
#pragma unroll
  for (int j = 0; j < 8; ++j) a[8 + j] = p[16 + j];
  return a;
}

// ---------------------------------------------------------------------------
// Pack a f32 row-major [Ksub, N] matrix into pre-swizzled f16 B-fragments.
// B layout: lane n (n=lane&15, sel=lane>>4) half j holds element
// (k = kt*32 + (j&8)*2 + sel*8 + (j&7), col = nt*16 + n) -> contiguous 32B per lane.
// ---------------------------------------------------------------------------
__global__ void pack_w(const float* __restrict__ src, half_t* __restrict__ dst,
                       int N, int NTtot, int nt_off, int kt_off, long total) {
  long gid = (long)blockIdx.x * blockDim.x + threadIdx.x;
  if (gid >= total) return;
  const int  j    = (int)(gid & 15);
  const int  lane = (int)((gid >> 4) & 31);
  const long rest = gid >> 9;
  const int  NTl  = N >> 4;
  const int  ntl  = (int)(rest % NTl);
  const int  ktl  = (int)(rest / NTl);
  const int  nn = lane & 15, sel = lane >> 4;
  const int  k = ktl * 32 + (j & 8) * 2 + sel * 8 + (j & 7);
  const int  c = ntl * 16 + nn;
  dst[(((long)(kt_off + ktl) * NTtot + (nt_off + ntl)) * 32 + lane) * 16 + j] =
      (half_t)src[(long)k * N + c];
}

// ---------------------------------------------------------------------------
// Phase 1: G = f16( x @ [W_fb|W_sb|W_ib] + [e_f|e_s|e_i] ), stored in packed
// C-fragment tiles: Gp[((rt*48 + ct)*32 + lane)*8 + r], rt = t*16 + bt.
// Block (t, bt) = 16 batch rows at fixed t; 8 waves x 6 n-tiles; K=512.
// Two independent accumulator chains (even/odd k-tiles) for ILP.
// ---------------------------------------------------------------------------
__global__ __launch_bounds__(256) void input_proj(
    const float* __restrict__ x, const float* __restrict__ e_f,
    const float* __restrict__ e_s, const float* __restrict__ e_i,
    const half_t* __restrict__ PB1, half_t* __restrict__ Gp) {
  const int lane = threadIdx.x & 31;
  const int wave = threadIdx.x >> 5;
  const int t    = blockIdx.x >> 4;      // 0..199
  const int bt   = blockIdx.x & 15;      // 0..15
  const int n = lane & 15;
  // x[b, t, k] : base = x + t*I_, row index b, row stride T_*I_
  const float* xbase = x + (long)t * I_;
  v16h A[16];
#pragma unroll
  for (int kt = 0; kt < 16; ++kt)
    A[kt] = a_frag_f32(xbase, (long)T_ * I_, bt * 16, kt * 32, lane);
  for (int q = 0; q < 6; ++q) {
    const int nt = wave * 6 + q;               // 0..47
    const int c  = nt * 16 + n;
    const float b = (c < 256) ? e_f[c] : ((c < 512) ? e_s[c - 256] : e_i[c - 512]);
    v8f acc0, acc1;
#pragma unroll
    for (int r = 0; r < 8; ++r) { acc0[r] = b; acc1[r] = 0.0f; }
#pragma unroll
    for (int kt = 0; kt < 16; kt += 2) {
      v16h B0 = *(const v16h*)(PB1 + ((long)((kt + 0) * 48 + nt) * 32 + lane) * 16);
      v16h B1 = *(const v16h*)(PB1 + ((long)((kt + 1) * 48 + nt) * 32 + lane) * 16);
      acc0 = __builtin_amdgcn_wmma_f32_16x16x32_f16(false, A[kt + 0], false, B0,
                                                    (short)0, acc0, false, false);
      acc1 = __builtin_amdgcn_wmma_f32_16x16x32_f16(false, A[kt + 1], false, B1,
                                                    (short)0, acc1, false, false);
    }
    v8h gh;
#pragma unroll
    for (int r = 0; r < 8; ++r) gh[r] = (_Float16)(acc0[r] + acc1[r]);
    *(v8h*)(Gp + (((long)blockIdx.x * 48 + nt) * 32 + lane) * 8) = gh;
  }
}

// ---------------------------------------------------------------------------
// Phase 2: sequential recurrence. 16 WGs (16 batch rows each) x 512 threads
// (16 waves = 16 n-tiles of H). h and a=h*sg tiles in LDS (f16); each wave
// keeps an f32 register copy of its own C-tile of h across steps.
// Hid written [T, B, H] via one cooperative b128 store per thread per step.
// Each gate uses two independent WMMA accumulator chains (depth 4 not 8).
// ---------------------------------------------------------------------------
__global__ __launch_bounds__(512) void gru_recurrence(
    const half_t* __restrict__ Gp, const half_t* __restrict__ PWf,
    const half_t* __restrict__ PWs, const half_t* __restrict__ PWi,
    half_t* __restrict__ Hid) {
  __shared__ __align__(16) half_t hbuf[2][16 * 256];
  __shared__ __align__(16) half_t abuf[16 * 256];
  const int lane = threadIdx.x & 31;
  const int nt   = threadIdx.x >> 5;   // n-tile 0..15
  const int mt   = blockIdx.x;         // batch tile 0..15
  const int n = lane & 15, sel = lane >> 4;

  for (int i = threadIdx.x; i < 16 * 256 / 8; i += 512)
    *(v8h*)(&hbuf[0][i * 8]) = (v8h)(_Float16)0.0f;
  __syncthreads();

  v8f hreg;                            // this wave's own h tile, f32
#pragma unroll
  for (int r = 0; r < 8; ++r) hreg[r] = 0.0f;

  int cur = 0;
  for (int t = 0; t < T_; ++t) {
    const long rt = (long)t * 16 + mt;
    // ---- U = h@W_fa + Gf ; V = h@W_sa + Gs ----
    v16h Ah[8];
#pragma unroll
    for (int kt = 0; kt < 8; ++kt) Ah[kt] = a_frag_f16(hbuf[cur], 256, 0, kt * 32, lane);
    v8f U0, U1, V0, V1;
    {
      v8h gf = *(const v8h*)(Gp + ((rt * 48 + (0 + nt)) * 32 + lane) * 8);
      v8h gs = *(const v8h*)(Gp + ((rt * 48 + (16 + nt)) * 32 + lane) * 8);
#pragma unroll
      for (int r = 0; r < 8; ++r) {
        U0[r] = (float)gf[r]; U1[r] = 0.0f;
        V0[r] = (float)gs[r]; V1[r] = 0.0f;
      }
    }
#pragma unroll
    for (int kt = 0; kt < 8; kt += 2) {
      v16h Bf0 = *(const v16h*)(PWf + ((long)((kt + 0) * 16 + nt) * 32 + lane) * 16);
      v16h Bf1 = *(const v16h*)(PWf + ((long)((kt + 1) * 16 + nt) * 32 + lane) * 16);
      v16h Bs0 = *(const v16h*)(PWs + ((long)((kt + 0) * 16 + nt) * 32 + lane) * 16);
      v16h Bs1 = *(const v16h*)(PWs + ((long)((kt + 1) * 16 + nt) * 32 + lane) * 16);
      U0 = __builtin_amdgcn_wmma_f32_16x16x32_f16(false, Ah[kt + 0], false, Bf0,
                                                  (short)0, U0, false, false);
      U1 = __builtin_amdgcn_wmma_f32_16x16x32_f16(false, Ah[kt + 1], false, Bf1,
                                                  (short)0, U1, false, false);
      V0 = __builtin_amdgcn_wmma_f32_16x16x32_f16(false, Ah[kt + 0], false, Bs0,
                                                  (short)0, V0, false, false);
      V1 = __builtin_amdgcn_wmma_f32_16x16x32_f16(false, Ah[kt + 1], false, Bs1,
                                                  (short)0, V1, false, false);
    }
    v8f fg;
#pragma unroll
    for (int r = 0; r < 8; ++r) {
      fg[r] = sigmoidf_(U0[r] + U1[r]);
      const float sg = sigmoidf_(V0[r] + V1[r]);
      abuf[(r + 8 * sel) * 256 + nt * 16 + n] = (half_t)(hreg[r] * sg);
    }
    __syncthreads();
    // ---- Wc = a@W_ia + Gi ; cg = tanh ; h update ----
    v8f W0, W1;
    {
      v8h gi = *(const v8h*)(Gp + ((rt * 48 + (32 + nt)) * 32 + lane) * 8);
#pragma unroll
      for (int r = 0; r < 8; ++r) { W0[r] = (float)gi[r]; W1[r] = 0.0f; }
    }
#pragma unroll
    for (int kt = 0; kt < 8; kt += 2) {
      v16h Aa0 = a_frag_f16(abuf, 256, 0, (kt + 0) * 32, lane);
      v16h Aa1 = a_frag_f16(abuf, 256, 0, (kt + 1) * 32, lane);
      v16h Bi0 = *(const v16h*)(PWi + ((long)((kt + 0) * 16 + nt) * 32 + lane) * 16);
      v16h Bi1 = *(const v16h*)(PWi + ((long)((kt + 1) * 16 + nt) * 32 + lane) * 16);
      W0 = __builtin_amdgcn_wmma_f32_16x16x32_f16(false, Aa0, false, Bi0,
                                                  (short)0, W0, false, false);
      W1 = __builtin_amdgcn_wmma_f32_16x16x32_f16(false, Aa1, false, Bi1,
                                                  (short)0, W1, false, false);
    }
    const int nxt = cur ^ 1;
#pragma unroll
    for (int r = 0; r < 8; ++r) {
      const float cg = tanhf(W0[r] + W1[r]);
      const float hn = hreg[r] * (1.0f - fg[r]) + fg[r] * cg;
      hreg[r] = hn;
      hbuf[nxt][(r + 8 * sel) * 256 + nt * 16 + n] = (half_t)hn;
    }
    __syncthreads();
    // cooperative write-out of h_new to Hid[T,B,H]: one b128 per thread
    {
      const long gbase = ((long)t * B_ + mt * 16) * H_;
      *(v8h*)(Hid + gbase + threadIdx.x * 8) =
          *(const v8h*)(&hbuf[nxt][threadIdx.x * 8]);
    }
    cur = nxt;
  }
}

// ---------------------------------------------------------------------------
// Phase 3: out = sigmoid( [hiddens | dt] @ [W_ba ; E] + e_o ),  K = 512.
// Block (t, bt): 16 batch rows at fixed t; Hid rows contiguous in [T,B,H];
// dt gathered from f[:, :, 1, mask_idx] and converted to f16 on the fly.
// ---------------------------------------------------------------------------
__global__ __launch_bounds__(256) void output_proj(
    const half_t* __restrict__ Hid, const float* __restrict__ f,
    const int* __restrict__ mask, const float* __restrict__ e_o,
    const half_t* __restrict__ PB2, float* __restrict__ out) {
  const int lane = threadIdx.x & 31;
  const int wave = threadIdx.x >> 5;
  const int t    = blockIdx.x >> 4;      // 0..199
  const int bt   = blockIdx.x & 15;      // 0..15
  const int n = lane & 15, sel = lane >> 4;
  v16h A[16];
  const half_t* hbase = Hid + ((long)t * B_ + bt * 16) * H_;
#pragma unroll
  for (int kt = 0; kt < 8; ++kt) A[kt] = a_frag_f16(hbase, H_, 0, kt * 32, lane);
#pragma unroll
  for (int kt = 0; kt < 8; ++kt) {
    const int m = lane & 15;
    const int b = bt * 16 + m;
    const long base = ((long)(b * T_ + t) * 2 + 1) * I_;   // f[b,t,1,:]
    const int k0 = kt * 32 + sel * 8;
    v16h a;
#pragma unroll
    for (int j = 0; j < 8; ++j) a[j]     = (_Float16)f[base + mask[k0 + j]];
#pragma unroll
    for (int j = 0; j < 8; ++j) a[8 + j] = (_Float16)f[base + mask[k0 + 16 + j]];
    A[8 + kt] = a;
  }
  for (int q = 0; q < 4; ++q) {
    const int nt = wave * 4 + q;           // 0..31
    const int c  = nt * 16 + n;
    const float b = e_o[c];
    v8f acc0, acc1;
#pragma unroll
    for (int r = 0; r < 8; ++r) { acc0[r] = b; acc1[r] = 0.0f; }
#pragma unroll
    for (int kt = 0; kt < 16; kt += 2) {
      v16h B0 = *(const v16h*)(PB2 + ((long)((kt + 0) * 32 + nt) * 32 + lane) * 16);
      v16h B1 = *(const v16h*)(PB2 + ((long)((kt + 1) * 32 + nt) * 32 + lane) * 16);
      acc0 = __builtin_amdgcn_wmma_f32_16x16x32_f16(false, A[kt + 0], false, B0,
                                                    (short)0, acc0, false, false);
      acc1 = __builtin_amdgcn_wmma_f32_16x16x32_f16(false, A[kt + 1], false, B1,
                                                    (short)0, acc1, false, false);
    }
#pragma unroll
    for (int r = 0; r < 8; ++r) {
      const int b_row = bt * 16 + r + 8 * sel;
      out[(long)(b_row * T_ + t) * I_ + c] = sigmoidf_(acc0[r] + acc1[r]);
    }
  }
}

// ---------------------------------------------------------------------------
extern "C" void kernel_launch(void* const* d_in, const int* in_sizes, int n_in,
                              void* d_out, int out_size, void* d_ws, size_t ws_size,
                              hipStream_t stream) {
  const float* x    = (const float*)d_in[0];
  const float* f    = (const float*)d_in[1];
  const int*   mask = (const int*)  d_in[2];
  const float* W_fb = (const float*)d_in[3];
  const float* W_fa = (const float*)d_in[4];
  const float* e_f  = (const float*)d_in[5];
  const float* W_ib = (const float*)d_in[6];
  const float* W_ia = (const float*)d_in[7];
  const float* e_i  = (const float*)d_in[8];
  const float* W_sb = (const float*)d_in[9];
  const float* W_sa = (const float*)d_in[10];
  const float* e_s  = (const float*)d_in[11];
  const float* W_ba = (const float*)d_in[12];
  const float* e_o  = (const float*)d_in[13];
  const float* E    = (const float*)d_in[14];

  char* ws = (char*)d_ws;
  half_t* PB1 = (half_t*)(ws + 0);          // 512x768 f16 packed  (786432 B)
  half_t* PWf = (half_t*)(ws + 786432);     // 256x256 f16 packed  (131072 B)
  half_t* PWs = (half_t*)(ws + 917504);
  half_t* PWi = (half_t*)(ws + 1048576);
  half_t* PB2 = (half_t*)(ws + 1179648);    // 512x512 f16 packed  (524288 B)
  half_t* Gp  = (half_t*)(ws + 1703936);    // packed C-tiles      (78643200 B)
  half_t* Hid = (half_t*)(ws + 80347136);   // [T,B,H] f16         (26214400 B)
  float*  out = (float*)d_out;

  // Pack weights into pre-swizzled WMMA B-fragment layout.
  pack_w<<<512, 256, 0, stream>>>(W_fb, PB1, 256, 48,  0, 0, 131072L);
  pack_w<<<512, 256, 0, stream>>>(W_sb, PB1, 256, 48, 16, 0, 131072L);
  pack_w<<<512, 256, 0, stream>>>(W_ib, PB1, 256, 48, 32, 0, 131072L);
  pack_w<<<256, 256, 0, stream>>>(W_fa, PWf, 256, 16,  0, 0, 65536L);
  pack_w<<<256, 256, 0, stream>>>(W_sa, PWs, 256, 16,  0, 0, 65536L);
  pack_w<<<256, 256, 0, stream>>>(W_ia, PWi, 256, 16,  0, 0, 65536L);
  pack_w<<<512, 256, 0, stream>>>(W_ba, PB2, 512, 32,  0, 0, 131072L);
  pack_w<<<512, 256, 0, stream>>>(E,    PB2, 512, 32,  0, 8, 131072L);

  input_proj    <<<NROW / 16, 256, 0, stream>>>(x, e_f, e_s, e_i, PB1, Gp);
  gru_recurrence<<<B_ / 16,   512, 0, stream>>>(Gp, PWf, PWs, PWi, Hid);
  output_proj   <<<NROW / 16, 256, 0, stream>>>(Hid, f, mask, e_o, PB2, out);
}
